// WIKG_11776800326092
// MI455X (gfx1250) — matrix operand order repaired and checked by reading, not compiled, not run on hardware
//
#include <hip/hip_runtime.h>
#include <hip/hip_bf16.h>
#include <math.h>

// ---------------- problem constants ----------------
#define BB       4
#define NN       4096
#define IN_DIM   1024
#define EMBED    512
#define ATTN     128
#define KNN      8
#define NCLS     2
#define ROWS     (BB*NN)        // 16384 flattened rows
#define EPS_LN   1e-5f
#define EPS_NORM 1e-12f

// ---------------- WMMA types ----------------
typedef __attribute__((ext_vector_type(16))) _Float16 v16h;
typedef __attribute__((ext_vector_type(8)))  float    v8f;

union FragH { v16h v; _Float16 h[16]; };

// A-matrix fragment, 16x32 f16:
//   lane L: m = L&15, half = L>>4
//   16 halves hold k = half*16 + {0..15}   (matches load_frag_bt)
__device__ inline v16h load_frag_a(const _Float16* base, int stride, int lane) {
  int m = lane & 15, half = lane >> 4;
  FragH f;
  const _Float16* p = base + (size_t)m * stride + half * 16;
#pragma unroll
  for (int e = 0; e < 16; ++e) f.h[e] = p[e];
  return f.v;
}

// B-matrix fragment, 32x16 f16, stored K-contiguous per column (i.e. B^T):
//   lane L: n = L&15, half = L>>4; 16 halves hold k = half*16 + {0..15}
__device__ inline v16h load_frag_bt(const _Float16* base, int stride, int lane) {
  int n = lane & 15, half = lane >> 4;
  FragH f;
  const _Float16* p = base + (size_t)n * stride + half * 16;
#pragma unroll
  for (int e = 0; e < 16; ++e) f.h[e] = p[e];
  return f.v;
}

template <int ACT>
__device__ inline float act_fn(float x) {
  if (ACT == 1) return tanhf(x);
  if (ACT == 2) return 1.0f / (1.0f + expf(-x));
  return x;
}

// ---------------------------------------------------------------
// Generic GEMM:  C[M,N] = act( A[M,K](f32) @ B[K,N](f32) + bias )
// 256 threads (8 waves), block tile 64x64, k-step 32.
// Software-pipelined: double-buffered LDS tiles; next k-tile's
// global loads are issued BEFORE the current tile's WMMAs so the
// VMEM latency overlaps matrix math; f32->f16 convert + LDS store
// happen after the WMMAs.
// Requires M%64==0, N%64==0, K%32==0 (true for all uses here).
// ---------------------------------------------------------------
template <int ACT>
__global__ void gemm_wmma_kernel(const float* __restrict__ A,
                                 const float* __restrict__ B,
                                 const float* __restrict__ bias,
                                 float* __restrict__ C,
                                 int M, int N, int K) {
  __shared__ __align__(16) _Float16 As[2][64 * 40];  // 64 m x 32 k (+8 pad)
  __shared__ __align__(16) _Float16 Bs[2][64 * 40];  // 64 n x 32 k (+8 pad), transposed

  const int t = threadIdx.x;
  const int wave = t >> 5, lane = t & 31;
  const int wr = wave >> 1, wc = wave & 1;
  const int m0 = blockIdx.y * 64;
  const int n0 = blockIdx.x * 64;

  float ra[8], rb[8];
  // prologue: fetch + stage k-tile 0
#pragma unroll
  for (int r = 0; r < 8; ++r) {
    int e = r * 256 + t;
    ra[r] = A[(size_t)(m0 + (e >> 5)) * K + (e & 31)];
    rb[r] = B[(size_t)(e >> 6) * N + n0 + (e & 63)];
  }
#pragma unroll
  for (int r = 0; r < 8; ++r) {
    int e = r * 256 + t;
    As[0][(e >> 5) * 40 + (e & 31)] = (_Float16)ra[r];
    Bs[0][(e & 63) * 40 + (e >> 6)] = (_Float16)rb[r];
  }

  v8f acc0 = {}, acc1 = {};
  int cur = 0;
  for (int k0 = 0; k0 < K; k0 += 32) {
    __syncthreads();                       // staged tile visible
    const bool more = (k0 + 32) < K;
    if (more) {                            // issue next tile's global loads now
#pragma unroll
      for (int r = 0; r < 8; ++r) {
        int e = r * 256 + t;
        ra[r] = A[(size_t)(m0 + (e >> 5)) * K + k0 + 32 + (e & 31)];
        rb[r] = B[(size_t)(k0 + 32 + (e >> 6)) * N + n0 + (e & 63)];
      }
    }
    v16h fa  = load_frag_a (&As[cur][(wr * 16) * 40], 40, lane);
    v16h fb0 = load_frag_bt(&Bs[cur][(wc * 32) * 40], 40, lane);
    v16h fb1 = load_frag_bt(&Bs[cur][(wc * 32 + 16) * 40], 40, lane);
    acc0 = __builtin_amdgcn_wmma_f32_16x16x32_f16(false, fa, false, fb0,
                                                  (short)0, acc0, false, false);
    acc1 = __builtin_amdgcn_wmma_f32_16x16x32_f16(false, fa, false, fb1,
                                                  (short)0, acc1, false, false);
    if (more) {                            // convert + store to alternate buffer
#pragma unroll
      for (int r = 0; r < 8; ++r) {
        int e = r * 256 + t;
        As[cur ^ 1][(e >> 5) * 40 + (e & 31)] = (_Float16)ra[r];
        Bs[cur ^ 1][(e & 63) * 40 + (e >> 6)] = (_Float16)rb[r];
      }
    }
    cur ^= 1;
  }

  // epilogue: C/D layout — lane L: n = L&15; vgpr r: m = r + 8*(L>>4)
  const int half = lane >> 4, n = lane & 15;
#pragma unroll
  for (int r = 0; r < 8; ++r) {
    int gm  = m0 + wr * 16 + r + 8 * half;
    int gn0 = n0 + wc * 32 + n;
    int gn1 = gn0 + 16;
    C[(size_t)gm * N + gn0] = act_fn<ACT>(acc0[r] + bias[gn0]);
    C[(size_t)gm * N + gn1] = act_fn<ACT>(acc1[r] + bias[gn1]);
  }
}

// ---------------------------------------------------------------
// Row L2-normalize h (f32, ROWS x EMBED) -> hn (f16). 1 wave / row.
// ---------------------------------------------------------------
__global__ void normalize_kernel(const float* __restrict__ h,
                                 _Float16* __restrict__ hn) {
  const int wave = threadIdx.x >> 5, lane = threadIdx.x & 31;
  const size_t row = (size_t)blockIdx.x * 8 + wave;
  const float* p = h + row * EMBED;
  float ss = 0.f;
#pragma unroll
  for (int e = 0; e < EMBED / 32; ++e) { float v = p[lane + 32 * e]; ss += v * v; }
#pragma unroll
  for (int off = 16; off > 0; off >>= 1) ss += __shfl_xor(ss, off, 32);
  const float inv = 1.0f / fmaxf(sqrtf(ss), EPS_NORM);
  _Float16* q = hn + row * EMBED;
#pragma unroll
  for (int e = 0; e < EMBED / 32; ++e)
    q[lane + 32 * e] = (_Float16)(p[lane + 32 * e] * inv);
}

// ---------------------------------------------------------------
// Fused cosine-sim + top-8 + softmax weights.
// Block = 128 threads (4 waves) owns a 16-row strip of one batch.
// A strip staged in LDS once; B fragments streamed from global hn.
// Inner k-loop is software-pipelined (next fragments loaded before
// the current WMMA issues) and the next j-tile is prefetched.
// ---------------------------------------------------------------
__global__ void simtopk_kernel(const _Float16* __restrict__ hn,
                               int* __restrict__ top_idx,
                               float* __restrict__ top_w) {
  __shared__ __align__(16) _Float16 Ast[16 * 520];   // 16 rows x 512 (+8 pad)
  __shared__ float simb[16 * 68];                    // 16 rows x 64 (+4 pad)

  const int t = threadIdx.x, wave = t >> 5, lane = t & 31;
  const int strip = blockIdx.x;                      // 0 .. B*(N/16)-1
  const int batch = strip >> 8;                      // N/16 == 256
  const int i0 = (strip & 255) * 16;
  const _Float16* hb = hn + (size_t)batch * NN * EMBED;

#pragma unroll 4
  for (int r = 0; r < 64; ++r) {                     // stage A strip
    int e = r * 128 + t;
    int row = e >> 9, col = e & 511;
    Ast[row * 520 + col] = hb[(size_t)(i0 + row) * EMBED + col];
  }
  __syncthreads();

  float tv[KNN]; int ti[KNN];
#pragma unroll
  for (int q = 0; q < KNN; ++q) { tv[q] = -2.0f; ti[q] = 0; }

  for (int j0 = 0; j0 < NN; j0 += 64) {
    const int jc = j0 + wave * 16;
    const _Float16* bbase = hb + (size_t)jc * EMBED;

    // prefetch next j-tile's B rows (gfx1250 global_prefetch)
    if (j0 + 64 < NN) {
      const _Float16* nb = hb + (size_t)(j0 + 64 + (lane & 15)) * EMBED
                              + (lane >> 4) * 256;
      __builtin_prefetch(nb, 0, 1);
    }

    v8f acc = {};
    v16h fa = load_frag_a(&Ast[0], 520, lane);
    v16h fb = load_frag_bt(bbase, EMBED, lane);
#pragma unroll 4
    for (int k0 = 0; k0 < EMBED - 32; k0 += 32) {    // pipelined: preload k+32
      v16h fa2 = load_frag_a(&Ast[k0 + 32], 520, lane);
      v16h fb2 = load_frag_bt(bbase + k0 + 32, EMBED, lane);
      acc = __builtin_amdgcn_wmma_f32_16x16x32_f16(false, fa, false, fb,
                                                   (short)0, acc, false, false);
      fa = fa2; fb = fb2;
    }
    acc = __builtin_amdgcn_wmma_f32_16x16x32_f16(false, fa, false, fb,
                                                 (short)0, acc, false, false);

    const int half = lane >> 4, n = lane & 15;
#pragma unroll
    for (int r = 0; r < 8; ++r) {
      int m = r + 8 * half;
      float v = acc[r];
      if (i0 + m == jc + n) v = -1.0f;               // self-loop suppressed
      simb[m * 68 + wave * 16 + n] = v;
    }
    __syncthreads();
    if (t < 16) {                                    // row owner merges 64 cands
      for (int c = 0; c < 64; ++c) {
        float v = simb[t * 68 + c];
        if (v > tv[KNN - 1]) {
          int p = KNN - 1;
          while (p > 0 && v > tv[p - 1]) { tv[p] = tv[p - 1]; ti[p] = ti[p - 1]; --p; }
          tv[p] = v; ti[p] = j0 + c;
        }
      }
    }
    __syncthreads();
  }

  if (t < 16) {                                      // softmax over kept sims
    float mx = tv[0], w[KNN], s = 0.f;
#pragma unroll
    for (int q = 0; q < KNN; ++q) { w[q] = expf(tv[q] - mx); s += w[q]; }
    float inv = 1.0f / s;
    size_t base = ((size_t)batch * NN + i0 + t) * KNN;
#pragma unroll
    for (int q = 0; q < KNN; ++q) { top_idx[base + q] = ti[q]; top_w[base + q] = w[q] * inv; }
  }
}

// ---------------------------------------------------------------
// h2 = LayerNorm( h + sum_k w_k * V[idx_k] ). One block / row.
// ---------------------------------------------------------------
__global__ void agg_ln_kernel(const float* __restrict__ h,
                              const float* __restrict__ V,
                              const int* __restrict__ top_idx,
                              const float* __restrict__ top_w,
                              const float* __restrict__ ln_g,
                              const float* __restrict__ ln_b,
                              float* __restrict__ h2) {
  __shared__ float red[8];
  __shared__ float stat;
  const int t = threadIdx.x, lane = t & 31, wv = t >> 5;
  const size_t row = blockIdx.x;
  const size_t vbase = (row >> 12) * (size_t)NN;     // batch base row for V

  int idx[KNN]; float w[KNN];
#pragma unroll
  for (int q = 0; q < KNN; ++q) { idx[q] = top_idx[row * KNN + q]; w[q] = top_w[row * KNN + q]; }

  float vals[2];
#pragma unroll
  for (int c = 0; c < 2; ++c) {
    int e = t + 256 * c;
    float a = h[row * EMBED + e];
#pragma unroll
    for (int q = 0; q < KNN; ++q) a += w[q] * V[(vbase + idx[q]) * EMBED + e];
    vals[c] = a;
  }
  float s = vals[0] + vals[1];
#pragma unroll
  for (int off = 16; off > 0; off >>= 1) s += __shfl_xor(s, off, 32);
  if (lane == 0) red[wv] = s;
  __syncthreads();
  if (t == 0) { float m = 0; for (int q = 0; q < 8; ++q) m += red[q]; stat = m * (1.0f / EMBED); }
  __syncthreads();
  const float mu = stat;
  __syncthreads();
  float d0 = vals[0] - mu, d1 = vals[1] - mu;
  s = d0 * d0 + d1 * d1;
#pragma unroll
  for (int off = 16; off > 0; off >>= 1) s += __shfl_xor(s, off, 32);
  if (lane == 0) red[wv] = s;
  __syncthreads();
  if (t == 0) { float m = 0; for (int q = 0; q < 8; ++q) m += red[q]; stat = m * (1.0f / EMBED); }
  __syncthreads();
  const float rstd = rsqrtf(stat + EPS_LN);
#pragma unroll
  for (int c = 0; c < 2; ++c) {
    int e = t + 256 * c;
    h2[row * EMBED + e] = (vals[c] - mu) * rstd * ln_g[e] + ln_b[e];
  }
}

// scores[i] = dot(AV[i]*AU[i], aw_w) + aw_b      (1 wave / row)
__global__ void score_kernel(const float* __restrict__ AV,
                             const float* __restrict__ AU,
                             const float* __restrict__ aw_w,
                             const float* __restrict__ aw_b,
                             float* __restrict__ scores) {
  const int wave = threadIdx.x >> 5, lane = threadIdx.x & 31;
  const size_t row = (size_t)blockIdx.x * 8 + wave;
  float s = 0.f;
#pragma unroll
  for (int e = 0; e < ATTN / 32; ++e) {
    int a = lane + 32 * e;
    s += AV[row * ATTN + a] * AU[row * ATTN + a] * aw_w[a];
  }
#pragma unroll
  for (int off = 16; off > 0; off >>= 1) s += __shfl_xor(s, off, 32);
  if (lane == 0) scores[row] = s + aw_b[0];
}

// softmax over N per batch. 1 block / batch.
__global__ void softmax_kernel(const float* __restrict__ scores,
                               float* __restrict__ Aout) {
  __shared__ float red[8];
  __shared__ float stat;
  const int t = threadIdx.x, lane = t & 31, wv = t >> 5;
  const float* s = scores + (size_t)blockIdx.x * NN;
  float* o = Aout + (size_t)blockIdx.x * NN;

  float mx = -1e30f;
  for (int i = t; i < NN; i += 256) mx = fmaxf(mx, s[i]);
#pragma unroll
  for (int off = 16; off > 0; off >>= 1) mx = fmaxf(mx, __shfl_xor(mx, off, 32));
  if (lane == 0) red[wv] = mx;
  __syncthreads();
  if (t == 0) { float m = red[0]; for (int q = 1; q < 8; ++q) m = fmaxf(m, red[q]); stat = m; }
  __syncthreads();
  const float m = stat;
  float sum = 0.f;
  for (int i = t; i < NN; i += 256) sum += expf(s[i] - m);
#pragma unroll
  for (int off = 16; off > 0; off >>= 1) sum += __shfl_xor(sum, off, 32);
  if (lane == 0) red[wv] = sum;
  __syncthreads();
  if (t == 0) { float ss = 0; for (int q = 0; q < 8; ++q) ss += red[q]; stat = ss; }
  __syncthreads();
  const float inv = 1.0f / stat;
  for (int i = t; i < NN; i += 256) o[i] = expf(s[i] - m) * inv;
}

// bag[b,e] = sum_i A[b,i] * h2[b,i,e].  grid (B*2), thread = one e.
__global__ void bag_kernel(const float* __restrict__ Aout,
                           const float* __restrict__ h2,
                           float* __restrict__ bag) {
  const int b = blockIdx.x >> 1;
  const int e = (blockIdx.x & 1) * 256 + threadIdx.x;
  const float* hb = h2 + (size_t)b * NN * EMBED;
  const float* ab = Aout + (size_t)b * NN;
  float s = 0.f;
  for (int i = 0; i < NN; ++i) s += ab[i] * hb[(size_t)i * EMBED + e];
  bag[b * EMBED + e] = s;
}

// out[b,c] = bag[b,:] @ cls_w[:,c] + cls_b[c]
__global__ void cls_kernel(const float* __restrict__ bag,
                           const float* __restrict__ cls_w,
                           const float* __restrict__ cls_b,
                           float* __restrict__ out) {
  const int t = threadIdx.x;
  if (t >= BB * NCLS) return;
  const int b = t >> 1, c = t & 1;
  float s = cls_b[c];
  for (int e = 0; e < EMBED; ++e) s += bag[b * EMBED + e] * cls_w[e * NCLS + c];
  out[b * NCLS + c] = s;
}

// ---------------------------------------------------------------
extern "C" void kernel_launch(void* const* d_in, const int* in_sizes, int n_in,
                              void* d_out, int out_size, void* d_ws, size_t ws_size,
                              hipStream_t stream) {
  (void)in_sizes; (void)n_in; (void)out_size; (void)ws_size;
  const float* x     = (const float*)d_in[0];
  const float* pe_w  = (const float*)d_in[1];
  const float* pe_b  = (const float*)d_in[2];
  const float* wv_w  = (const float*)d_in[3];
  const float* wv_b  = (const float*)d_in[4];
  const float* ln_g  = (const float*)d_in[5];
  const float* ln_b  = (const float*)d_in[6];
  const float* av_w  = (const float*)d_in[7];
  const float* av_b  = (const float*)d_in[8];
  const float* au_w  = (const float*)d_in[9];
  const float* au_b  = (const float*)d_in[10];
  const float* aw_w  = (const float*)d_in[11];
  const float* aw_b  = (const float*)d_in[12];
  const float* cls_w = (const float*)d_in[13];
  const float* cls_b = (const float*)d_in[14];
  float* out = (float*)d_out;

  char* ws = (char*)d_ws;
  size_t off = 0;
  auto take = [&](size_t bytes) { char* p = ws + off; off += (bytes + 255) & ~(size_t)255; return p; };
  float*    h       = (float*)   take((size_t)ROWS * EMBED * 4);
  float*    V       = (float*)   take((size_t)ROWS * EMBED * 4);
  float*    h2      = (float*)   take((size_t)ROWS * EMBED * 4);
  _Float16* hn      = (_Float16*)take((size_t)ROWS * EMBED * 2);
  float*    AV      = (float*)   take((size_t)ROWS * ATTN * 4);
  float*    AU      = (float*)   take((size_t)ROWS * ATTN * 4);
  int*      top_idx = (int*)     take((size_t)ROWS * KNN * 4);
  float*    top_w   = (float*)   take((size_t)ROWS * KNN * 4);
  float*    scores  = (float*)   take((size_t)ROWS * 4);
  float*    Asm     = (float*)   take((size_t)ROWS * 4);
  float*    bag     = (float*)   take((size_t)BB * EMBED * 4);

  // 1) h = x @ pe_w + pe_b
  gemm_wmma_kernel<0><<<dim3(EMBED / 64, ROWS / 64), 256, 0, stream>>>(
      x, pe_w, pe_b, h, ROWS, EMBED, IN_DIM);
  // 2) hn = normalize(h)
  normalize_kernel<<<ROWS / 8, 256, 0, stream>>>(h, hn);
  // 3) fused sim + top-k + softmax weights
  simtopk_kernel<<<BB * (NN / 16), 128, 0, stream>>>(hn, top_idx, top_w);
  // 4) V = h @ wv_w + wv_b
  gemm_wmma_kernel<0><<<dim3(EMBED / 64, ROWS / 64), 256, 0, stream>>>(
      h, wv_w, wv_b, V, ROWS, EMBED, EMBED);
  // 5) h2 = LN(h + adj @ V)  (sparse 8-way gather)
  agg_ln_kernel<<<ROWS, 256, 0, stream>>>(h, V, top_idx, top_w, ln_g, ln_b, h2);
  // 6) gated attention projections
  gemm_wmma_kernel<1><<<dim3(ATTN / 64, ROWS / 64), 256, 0, stream>>>(
      h2, av_w, av_b, AV, ROWS, ATTN, EMBED);
  gemm_wmma_kernel<2><<<dim3(ATTN / 64, ROWS / 64), 256, 0, stream>>>(
      h2, au_w, au_b, AU, ROWS, ATTN, EMBED);
  // 7) scores, softmax over N, bag, classifier
  score_kernel<<<ROWS / 8, 256, 0, stream>>>(AV, AU, aw_w, aw_b, scores);
  softmax_kernel<<<BB, 256, 0, stream>>>(scores, Asm);
  bag_kernel<<<BB * 2, 256, 0, stream>>>(Asm, h2, bag);
  cls_kernel<<<1, 64, 0, stream>>>(bag, cls_w, cls_b, out);
}
